// Mamba2Mixer_67078799229203
// MI455X (gfx1250) — compile-verified, matrix operand order checked by reading, not run on hardware
//
#include <hip/hip_runtime.h>
#include <hip/hip_bf16.h>

#define L_SEQ     8192
#define DMODEL    1024
#define DINNER    2048
#define NHEADS    32
#define HEADDIM   64
#define DSTATE    128
#define CONVDIM   2304
#define DINPROJ   4384
#define DINPROJ_P 4480      // padded to a multiple of 128 for guard-free GEMM
#define LN_EPS    1e-5f
#define KSPLIT    16

typedef __bf16 bf16_t;
typedef __attribute__((ext_vector_type(16))) __bf16 v16bf;
typedef __attribute__((ext_vector_type(8)))  __bf16 v8bf;
typedef __attribute__((ext_vector_type(8)))  float  v8f;

// ---- helpers -------------------------------------------------------------

__device__ __forceinline__ bf16_t f2bf(float f) {
  union { float f; unsigned u; } v; v.f = f;
  unsigned r = v.u + 0x7FFFu + ((v.u >> 16) & 1u);   // RNE
  unsigned short h = (unsigned short)(r >> 16);
  return __builtin_bit_cast(bf16_t, h);
}

// Load one lane's 16 bf16 A/B-operand elements for v_wmma_f32_16x16x32_bf16.
// Per ISA layout: lanes 0-15 hold K={0..7, 16..23}, lanes 16-31 K={8..15, 24..31};
// caller folds the +8 lane-group offset into the pointer. Two 16B loads.
__device__ __forceinline__ v16bf ld_row16(const bf16_t* __restrict__ p) {
  v8bf lo = *(const v8bf*)(p);
  v8bf hi = *(const v8bf*)(p + 16);
  v16bf r;
#pragma unroll
  for (int i = 0; i < 8; ++i) { r[i] = lo[i]; r[i + 8] = hi[i]; }
  return r;
}

__device__ __forceinline__ v8f wmma_bf16(v16bf a, v16bf b, v8f c) {
  return __builtin_amdgcn_wmma_f32_16x16x32_bf16(false, a, false, b, (short)0, c,
                                                 false, false);
}

// ---- elementwise utility kernels ----------------------------------------

__global__ void f32_to_bf16_kernel(const float* __restrict__ in,
                                   bf16_t* __restrict__ out, int n) {
  int i = blockIdx.x * blockDim.x + threadIdx.x;
  if (i < n) out[i] = f2bf(in[i]);
}

__global__ void zero_f32_kernel(float* __restrict__ p, int n) {
  int i = blockIdx.x * blockDim.x + threadIdx.x;
  if (i < n) p[i] = 0.0f;
}

// ---- generic NT GEMM: C[M,N] = A[M,K] * B[N,K]^T, bf16 in / f32 out ------
// Block: 256 threads (8 waves), wave grid 4(M) x 2(N), wave tile 16M x 64N,
// block tile 64M x 128N. Requirements: K % 64 == 0, N % 128 == 0, M % 64 == 0.
// Software-pipelined: operand set for K-step k+32 is in flight while the
// WMMAs for K-step k execute (register double-buffering, no EXEC branching).

__global__ __launch_bounds__(256) void gemm_nt_bf16(
    const bf16_t* __restrict__ A, const bf16_t* __restrict__ B,
    float* __restrict__ C, int N, int K) {
  const int lane  = threadIdx.x & 31;
  const int wave  = threadIdx.x >> 5;
  const int m0    = blockIdx.y * 64 + (wave >> 1) * 16;
  const int n0    = blockIdx.x * 128 + (wave & 1) * 64;
  const int lgrp  = (lane >> 4) << 3;   // 0 or 8 (K lane-group offset)
  const int lr    = lane & 15;

  const bf16_t* arow  = A + (size_t)(m0 + lr) * K + lgrp;
  const bf16_t* brow0 = B + (size_t)(n0 +  0 + lr) * K + lgrp;
  const bf16_t* brow1 = B + (size_t)(n0 + 16 + lr) * K + lgrp;
  const bf16_t* brow2 = B + (size_t)(n0 + 32 + lr) * K + lgrp;
  const bf16_t* brow3 = B + (size_t)(n0 + 48 + lr) * K + lgrp;

  v8f acc0 = {}, acc1 = {}, acc2 = {}, acc3 = {};

  // stage A: operands for K-step k0; stage B: operands for K-step k0+32
  v16bf aA  = ld_row16(arow);
  v16bf bA0 = ld_row16(brow0);
  v16bf bA1 = ld_row16(brow1);
  v16bf bA2 = ld_row16(brow2);
  v16bf bA3 = ld_row16(brow3);

  for (int k0 = 0; k0 < K - 64; k0 += 64) {
    v16bf aB  = ld_row16(arow  + k0 + 32);
    v16bf bB0 = ld_row16(brow0 + k0 + 32);
    v16bf bB1 = ld_row16(brow1 + k0 + 32);
    v16bf bB2 = ld_row16(brow2 + k0 + 32);
    v16bf bB3 = ld_row16(brow3 + k0 + 32);
    acc0 = wmma_bf16(aA, bA0, acc0);
    acc1 = wmma_bf16(aA, bA1, acc1);
    acc2 = wmma_bf16(aA, bA2, acc2);
    acc3 = wmma_bf16(aA, bA3, acc3);
    aA  = ld_row16(arow  + k0 + 64);
    bA0 = ld_row16(brow0 + k0 + 64);
    bA1 = ld_row16(brow1 + k0 + 64);
    bA2 = ld_row16(brow2 + k0 + 64);
    bA3 = ld_row16(brow3 + k0 + 64);
    acc0 = wmma_bf16(aB, bB0, acc0);
    acc1 = wmma_bf16(aB, bB1, acc1);
    acc2 = wmma_bf16(aB, bB2, acc2);
    acc3 = wmma_bf16(aB, bB3, acc3);
  }
  {
    v16bf aB  = ld_row16(arow  + K - 32);
    v16bf bB0 = ld_row16(brow0 + K - 32);
    v16bf bB1 = ld_row16(brow1 + K - 32);
    v16bf bB2 = ld_row16(brow2 + K - 32);
    v16bf bB3 = ld_row16(brow3 + K - 32);
    acc0 = wmma_bf16(aA, bA0, acc0);
    acc1 = wmma_bf16(aA, bA1, acc1);
    acc2 = wmma_bf16(aA, bA2, acc2);
    acc3 = wmma_bf16(aA, bA3, acc3);
    acc0 = wmma_bf16(aB, bB0, acc0);
    acc1 = wmma_bf16(aB, bB1, acc1);
    acc2 = wmma_bf16(aB, bB2, acc2);
    acc3 = wmma_bf16(aB, bB3, acc3);
  }

  float* crow = C + (size_t)(m0 + lgrp) * N + n0 + lr;
#pragma unroll
  for (int r = 0; r < 8; ++r) {
    float* cr = crow + (size_t)r * N;
    cr[0]  = acc0[r];
    cr[16] = acc1[r];
    cr[32] = acc2[r];
    cr[48] = acc3[r];
  }
}

// ---- dt = softplus(raw + bias); coef = dt * exp(dt * -exp(A_log)) --------

__global__ void dt_coef_kernel(const float* __restrict__ zxb,
                               const float* __restrict__ dt_bias,
                               const float* __restrict__ A_log,
                               float* __restrict__ coef, int n) {
  int i = blockIdx.x * blockDim.x + threadIdx.x;
  if (i >= n) return;
  int l = i / NHEADS, h = i % NHEADS;
  float raw = zxb[(size_t)l * DINPROJ_P + (DINNER + CONVDIM) + h] + dt_bias[h];
  float dt  = raw > 20.f ? raw : log1pf(__expf(raw));
  float a   = __expf(-__expf(A_log[h]) * dt);
  coef[i]   = dt * a;
}

// ---- depthwise conv(k=4,SAME pad 1/2) + SiLU + split/transpose -----------
// Produces: xssm f32 (L x 2048), AVt bf16 (2048 x L) = (dt*a*x_ssm)^T,
//           Bt bf16 (128 x L) = B^T, Cb bf16 (L x 128).

__global__ void conv_split_kernel(const float* __restrict__ zxb,
                                  const float* __restrict__ W_conv,
                                  const float* __restrict__ b_conv,
                                  const float* __restrict__ coef,
                                  float*  __restrict__ xssm,
                                  bf16_t* __restrict__ AVt,
                                  bf16_t* __restrict__ Bt,
                                  bf16_t* __restrict__ Cb) {
  int idx = blockIdx.x * blockDim.x + threadIdx.x;
  if (idx >= L_SEQ * CONVDIM) return;
  int l = idx / CONVDIM, c = idx % CONVDIM;
  const float* col = zxb + DINNER + c;
  const float* w   = W_conv + c * 4;
  float acc = b_conv[c];
#pragma unroll
  for (int t = 0; t < 4; ++t) {
    int ls = l + t - 1;
    if (ls >= 0 && ls < L_SEQ) acc += w[t] * col[(size_t)ls * DINPROJ_P];
  }
  float v = acc / (1.f + __expf(-acc));           // SiLU
  if (c < DINNER) {
    int h = c >> 6;
    xssm[(size_t)l * DINNER + c] = v;
    AVt[(size_t)c * L_SEQ + l]   = f2bf(v * coef[l * NHEADS + h]);
  } else if (c < DINNER + DSTATE) {
    Bt[(size_t)(c - DINNER) * L_SEQ + l] = f2bf(v);
  } else {
    Cb[(size_t)l * DSTATE + (c - DINNER - DSTATE)] = f2bf(v);
  }
}

// ---- H[h] = B^T @ (a*dt*x_ssm)_h, split-K with fp32 atomics --------------
// Per block: one head, one K-slice of 512. 8 waves cover 8 n-tiles (128/16),
// each wave does the full 64-wide p dimension (4 N tiles). Output stored
// transposed: Ht[h][p][n] so the y-GEMM B operand is K(n)-contiguous.
// Same software-pipelined structure as gemm_nt_bf16.

__global__ __launch_bounds__(256) void h_gemm_kernel(
    const bf16_t* __restrict__ Bt, const bf16_t* __restrict__ AVt,
    float* __restrict__ Ht) {
  const int h     = blockIdx.y;
  const int kbase = blockIdx.x * (L_SEQ / KSPLIT);
  const int kend  = kbase + (L_SEQ / KSPLIT);
  const int lane  = threadIdx.x & 31;
  const int wave  = threadIdx.x >> 5;
  const int m0    = wave * 16;                       // n-dim of H
  const int lgrp  = (lane >> 4) << 3;
  const int lr    = lane & 15;
  const bf16_t* arow  = Bt + (size_t)(m0 + lr) * L_SEQ + lgrp;
  const bf16_t* brow0 = AVt + (size_t)(h * HEADDIM + lr +  0) * L_SEQ + lgrp;
  const bf16_t* brow1 = AVt + (size_t)(h * HEADDIM + lr + 16) * L_SEQ + lgrp;
  const bf16_t* brow2 = AVt + (size_t)(h * HEADDIM + lr + 32) * L_SEQ + lgrp;
  const bf16_t* brow3 = AVt + (size_t)(h * HEADDIM + lr + 48) * L_SEQ + lgrp;

  v8f acc0 = {}, acc1 = {}, acc2 = {}, acc3 = {};

  v16bf aA  = ld_row16(arow  + kbase);
  v16bf bA0 = ld_row16(brow0 + kbase);
  v16bf bA1 = ld_row16(brow1 + kbase);
  v16bf bA2 = ld_row16(brow2 + kbase);
  v16bf bA3 = ld_row16(brow3 + kbase);

  for (int k0 = kbase; k0 < kend - 64; k0 += 64) {
    v16bf aB  = ld_row16(arow  + k0 + 32);
    v16bf bB0 = ld_row16(brow0 + k0 + 32);
    v16bf bB1 = ld_row16(brow1 + k0 + 32);
    v16bf bB2 = ld_row16(brow2 + k0 + 32);
    v16bf bB3 = ld_row16(brow3 + k0 + 32);
    acc0 = wmma_bf16(aA, bA0, acc0);
    acc1 = wmma_bf16(aA, bA1, acc1);
    acc2 = wmma_bf16(aA, bA2, acc2);
    acc3 = wmma_bf16(aA, bA3, acc3);
    aA  = ld_row16(arow  + k0 + 64);
    bA0 = ld_row16(brow0 + k0 + 64);
    bA1 = ld_row16(brow1 + k0 + 64);
    bA2 = ld_row16(brow2 + k0 + 64);
    bA3 = ld_row16(brow3 + k0 + 64);
    acc0 = wmma_bf16(aB, bB0, acc0);
    acc1 = wmma_bf16(aB, bB1, acc1);
    acc2 = wmma_bf16(aB, bB2, acc2);
    acc3 = wmma_bf16(aB, bB3, acc3);
  }
  {
    v16bf aB  = ld_row16(arow  + kend - 32);
    v16bf bB0 = ld_row16(brow0 + kend - 32);
    v16bf bB1 = ld_row16(brow1 + kend - 32);
    v16bf bB2 = ld_row16(brow2 + kend - 32);
    v16bf bB3 = ld_row16(brow3 + kend - 32);
    acc0 = wmma_bf16(aA, bA0, acc0);
    acc1 = wmma_bf16(aA, bA1, acc1);
    acc2 = wmma_bf16(aA, bA2, acc2);
    acc3 = wmma_bf16(aA, bA3, acc3);
    acc0 = wmma_bf16(aB, bB0, acc0);
    acc1 = wmma_bf16(aB, bB1, acc1);
    acc2 = wmma_bf16(aB, bB2, acc2);
    acc3 = wmma_bf16(aB, bB3, acc3);
  }

  float* Hh = Ht + (size_t)h * HEADDIM * DSTATE;     // [p][n]
#pragma unroll
  for (int r = 0; r < 8; ++r) {
    const int n = m0 + lgrp + r;
    atomicAdd(&Hh[(size_t)(lr +  0) * DSTATE + n], acc0[r]);
    atomicAdd(&Hh[(size_t)(lr + 16) * DSTATE + n], acc1[r]);
    atomicAdd(&Hh[(size_t)(lr + 32) * DSTATE + n], acc2[r]);
    atomicAdd(&Hh[(size_t)(lr + 48) * DSTATE + n], acc3[r]);
  }
}

// ---- y[l,h,p] = C[l,:] @ Ht[h][p,:] + D[h]*x_ssm[l,h,p] ------------------
// Block: 64 seq rows x one head (64 cols); 8 waves as 4M x 2N, K = 128.

__global__ __launch_bounds__(256) void y_gemm_kernel(
    const bf16_t* __restrict__ Cb, const bf16_t* __restrict__ Htb,
    const float* __restrict__ xssm, const float* __restrict__ Dp,
    float* __restrict__ y) {
  const int h     = blockIdx.y;
  const int l0    = blockIdx.x * 64;
  const int lane  = threadIdx.x & 31;
  const int wave  = threadIdx.x >> 5;
  const int m0    = l0 + (wave >> 1) * 16;
  const int n0    = (wave & 1) * 32;
  const int lgrp  = (lane >> 4) << 3;
  const int lr    = lane & 15;
  const bf16_t* arow = Cb  + (size_t)(m0 + lr) * DSTATE + lgrp;
  const bf16_t* bb   = Htb + (size_t)h * HEADDIM * DSTATE
                           + (size_t)(n0 + lr) * DSTATE + lgrp;

  v8f acc0 = {}, acc1 = {};
#pragma unroll
  for (int k0 = 0; k0 < DSTATE; k0 += 32) {
    v16bf a  = ld_row16(arow + k0);
    v16bf b0 = ld_row16(bb + k0);
    v16bf b1 = ld_row16(bb + (size_t)16 * DSTATE + k0);
    acc0 = wmma_bf16(a, b0, acc0);
    acc1 = wmma_bf16(a, b1, acc1);
  }
  const float Dh = Dp[h];
#pragma unroll
  for (int r = 0; r < 8; ++r) {
    const int l = m0 + lgrp + r;
    const size_t o = (size_t)l * DINNER + h * HEADDIM + n0 + lr;
    y[o]      = acc0[r] + Dh * xssm[o];
    y[o + 16] = acc1[r] + Dh * xssm[o + 16];
  }
}

// ---- LayerNorm over 2048 + multiply by z gate, emit bf16 -----------------

__global__ __launch_bounds__(256) void ln_gate_kernel(
    const float* __restrict__ y, const float* __restrict__ zxb,
    const float* __restrict__ ln_w, const float* __restrict__ ln_b,
    bf16_t* __restrict__ yg) {
  __shared__ float red[256];
  const int l = blockIdx.x, tid = threadIdx.x;
  const float* row = y + (size_t)l * DINNER;
  float s = 0.f, ss = 0.f;
  for (int j = tid; j < DINNER; j += 256) { float v = row[j]; s += v; ss += v * v; }
  red[tid] = s; __syncthreads();
  for (int st = 128; st > 0; st >>= 1) { if (tid < st) red[tid] += red[tid + st]; __syncthreads(); }
  const float mean = red[0] / DINNER;
  __syncthreads();
  red[tid] = ss; __syncthreads();
  for (int st = 128; st > 0; st >>= 1) { if (tid < st) red[tid] += red[tid + st]; __syncthreads(); }
  const float var  = red[0] / DINNER - mean * mean;
  const float rstd = rsqrtf(var + LN_EPS);
  const float* zr = zxb + (size_t)l * DINPROJ_P;   // z = first 2048 cols
  for (int j = tid; j < DINNER; j += 256) {
    float v = (row[j] - mean) * rstd * ln_w[j] + ln_b[j];
    yg[(size_t)l * DINNER + j] = f2bf(v * zr[j]);
  }
}

// ---- host-side orchestration ---------------------------------------------

extern "C" void kernel_launch(void* const* d_in, const int* in_sizes, int n_in,
                              void* d_out, int out_size, void* d_ws, size_t ws_size,
                              hipStream_t stream) {
  const float* x       = (const float*)d_in[0];
  const float* W_in    = (const float*)d_in[1];
  const float* W_conv  = (const float*)d_in[2];
  const float* b_conv  = (const float*)d_in[3];
  const float* dt_bias = (const float*)d_in[4];
  const float* A_log   = (const float*)d_in[5];
  const float* D_par   = (const float*)d_in[6];
  const float* ln_w    = (const float*)d_in[7];
  const float* ln_b    = (const float*)d_in[8];
  const float* W_out   = (const float*)d_in[9];
  float* out = (float*)d_out;

  char* ws = (char*)d_ws;
  size_t off = 0;
  auto alloc = [&](size_t bytes) -> char* {
    char* p = ws + off;
    off += (bytes + 255) & ~(size_t)255;
    return p;
  };
  bf16_t* xb    = (bf16_t*)alloc((size_t)L_SEQ * DMODEL * 2);
  bf16_t* Winb  = (bf16_t*)alloc((size_t)DINPROJ_P * DMODEL * 2);  // rows 4384.. are pad
  bf16_t* Woutb = (bf16_t*)alloc((size_t)DMODEL * DINNER * 2);
  float*  zxb   = (float*) alloc((size_t)L_SEQ * DINPROJ_P * 4);   // padded row stride
  float*  coef  = (float*) alloc((size_t)L_SEQ * NHEADS * 4);
  float*  xssm  = (float*) alloc((size_t)L_SEQ * DINNER * 4);
  bf16_t* AVt   = (bf16_t*)alloc((size_t)DINNER * L_SEQ * 2);
  bf16_t* Bt    = (bf16_t*)alloc((size_t)DSTATE * L_SEQ * 2);
  bf16_t* Cb    = (bf16_t*)alloc((size_t)L_SEQ * DSTATE * 2);
  float*  Ht    = (float*) alloc((size_t)NHEADS * HEADDIM * DSTATE * 4);
  bf16_t* Htb   = (bf16_t*)alloc((size_t)NHEADS * HEADDIM * DSTATE * 2);
  float*  yb    = (float*) alloc((size_t)L_SEQ * DINNER * 4);
  bf16_t* yg    = (bf16_t*)alloc((size_t)L_SEQ * DINNER * 2);
  (void)ws_size; (void)in_sizes; (void)n_in; (void)out_size;

  // bf16 conversions of GEMM operands (pad rows of Winb are never consumed
  // downstream: the extra output columns they produce are simply unread)
  int n1 = L_SEQ * DMODEL;
  f32_to_bf16_kernel<<<(n1 + 255) / 256, 256, 0, stream>>>(x, xb, n1);
  int n2 = DINPROJ * DMODEL;
  f32_to_bf16_kernel<<<(n2 + 255) / 256, 256, 0, stream>>>(W_in, Winb, n2);
  int n3 = DMODEL * DINNER;
  f32_to_bf16_kernel<<<(n3 + 255) / 256, 256, 0, stream>>>(W_out, Woutb, n3);

  // 1) in-projection: zxb = x @ W_in^T   (8192 x 4480 padded)
  dim3 g1(DINPROJ_P / 128, L_SEQ / 64);
  gemm_nt_bf16<<<g1, 256, 0, stream>>>(xb, Winb, zxb, DINPROJ_P, DMODEL);

  // 2) dt/decay coefficients
  int nc = L_SEQ * NHEADS;
  dt_coef_kernel<<<(nc + 255) / 256, 256, 0, stream>>>(zxb, dt_bias, A_log, coef, nc);

  // 3) depthwise conv + SiLU + operand staging
  int np = L_SEQ * CONVDIM;
  conv_split_kernel<<<(np + 255) / 256, 256, 0, stream>>>(zxb, W_conv, b_conv, coef,
                                                          xssm, AVt, Bt, Cb);

  // 4) H = sum_l a*B (x) dt*x_ssm  (split-K over the 8192 sequence)
  int nh = NHEADS * HEADDIM * DSTATE;
  zero_f32_kernel<<<(nh + 255) / 256, 256, 0, stream>>>(Ht, nh);
  dim3 g2(KSPLIT, NHEADS);
  h_gemm_kernel<<<g2, 256, 0, stream>>>(Bt, AVt, Ht);
  f32_to_bf16_kernel<<<(nh + 255) / 256, 256, 0, stream>>>(Ht, Htb, nh);

  // 5) y = C @ H + D * x_ssm
  dim3 g3(L_SEQ / 64, NHEADS);
  y_gemm_kernel<<<g3, 256, 0, stream>>>(Cb, Htb, xssm, D_par, yb);

  // 6) LayerNorm + z gate
  ln_gate_kernel<<<L_SEQ, 256, 0, stream>>>(yb, zxb, ln_w, ln_b, yg);

  // 7) out-projection: out = yg @ W_out^T  (8192 x 1024)
  dim3 g4(DMODEL / 128, L_SEQ / 64);
  gemm_nt_bf16<<<g4, 256, 0, stream>>>(yg, Woutb, out, DMODEL, DINNER);
}
